// MultiHeadAttention_56006373540301
// MI455X (gfx1250) — matrix operand order repaired and checked by reading, not compiled, not run on hardware
//
#include <hip/hip_runtime.h>

typedef _Float16 f16;
typedef __attribute__((ext_vector_type(16))) _Float16 v16h;
typedef __attribute__((ext_vector_type(8)))  _Float16 v8h;
typedef __attribute__((ext_vector_type(4)))  _Float16 v4h;
typedef __attribute__((ext_vector_type(8)))  float    v8f;

constexpr int B_  = 2;
constexpr int T_  = 2048;
constexpr int C_  = 1024;
constexpr int H_  = 16;
constexpr int HS_ = 64;

__device__ __forceinline__ v8f wmma16x16x32(v16h a, v16h b, v8f c) {
    return __builtin_amdgcn_wmma_f32_16x16x32_f16(
        false, a, false, b, (short)0, c, false, false);
}
__device__ __forceinline__ v8h ld8(const f16* p) { return *(const v8h*)p; }
__device__ __forceinline__ v16h cat16(v8h lo, v8h hi) {
    return __builtin_shufflevector(lo, hi, 0,1,2,3,4,5,6,7,8,9,10,11,12,13,14,15);
}
__device__ __forceinline__ float rowmax16(float v) {
    v = fmaxf(v, __shfl_xor(v, 1));
    v = fmaxf(v, __shfl_xor(v, 2));
    v = fmaxf(v, __shfl_xor(v, 4));
    v = fmaxf(v, __shfl_xor(v, 8));
    return v;
}
__device__ __forceinline__ float rowsum16(float v) {
    v += __shfl_xor(v, 1);
    v += __shfl_xor(v, 2);
    v += __shfl_xor(v, 4);
    v += __shfl_xor(v, 8);
    return v;
}

// ---------------------------------------------------------------------------
// Pack pass: one-time f32 -> f16 conversion + weight transposes so all hot
// GEMM loops use contiguous 128-bit f16 loads with zero in-loop conversions.
// ---------------------------------------------------------------------------
__global__ __launch_bounds__(256)
void pack_x_kernel(const float* __restrict__ x, f16* __restrict__ xh) {
    const int i = blockIdx.x * 256 + threadIdx.x;       // one float4 each
    float4 f = ((const float4*)x)[i];
    v4h h; h[0] = (f16)f.x; h[1] = (f16)f.y; h[2] = (f16)f.z; h[3] = (f16)f.w;
    ((v4h*)xh)[i] = h;
}

// wq/wk/wv [H,C,HS] f32  ->  wt{q,k,v} [H,HS,C] f16   (B^T, row-major)
__global__ __launch_bounds__(256)
void pack_wt_kernel(const float* __restrict__ wq, const float* __restrict__ wk,
                    const float* __restrict__ wv,
                    f16* __restrict__ wtq, f16* __restrict__ wtk,
                    f16* __restrict__ wtv) {
    int id = blockIdx.x * 256 + threadIdx.x;
    const int n = id % HS_;  id /= HS_;
    const int k = id % C_;   id /= C_;
    const int h = id % H_;
    const int m = id / H_;
    const float* __restrict__ w  = (m == 0) ? wq : (m == 1) ? wk : wv;
    f16*         __restrict__ wt = (m == 0) ? wtq : (m == 1) ? wtk : wtv;
    wt[((size_t)h * HS_ + n) * C_ + k] = (f16)w[((size_t)h * C_ + k) * HS_ + n];
}

// wproj [K=1024, C=1024] f32 -> wpT [C, K] f16
__global__ __launch_bounds__(256)
void pack_wp_kernel(const float* __restrict__ wproj, f16* __restrict__ wpT) {
    const int id = blockIdx.x * 256 + threadIdx.x;
    const int n = id % C_;
    const int k = id / C_;
    wpT[(size_t)n * C_ + k] = (f16)wproj[(size_t)k * C_ + n];
}

// ---------------------------------------------------------------------------
// Kernel 1: QKV projection.  One wave computes a 32x64 tile for one
// (matrix, b, h, 32-row tile): 8 WMMA per 32-wide K-chunk, B reused 2x.
// q,k stored [B,H,T,HS]; v stored transposed [B,H,HS,T] (packed stores).
// ---------------------------------------------------------------------------
__global__ __launch_bounds__(128)
void qkv_proj_kernel(const f16* __restrict__ xh,
                     const f16* __restrict__ wtq, const f16* __restrict__ wtk,
                     const f16* __restrict__ wtv,
                     f16* __restrict__ qs, f16* __restrict__ ks,
                     f16* __restrict__ vt) {
    const int lane = threadIdx.x & 31;
    int wave = blockIdx.x * (blockDim.x >> 5) + (threadIdx.x >> 5);
    const int tt = wave % (T_ / 32);  wave /= (T_ / 32);
    const int h  = wave % H_;         wave /= H_;
    const int b  = wave % B_;
    const int m  = wave / B_;                    // 0=q, 1=k, 2=v

    const f16* __restrict__ wt = (m == 0) ? wtq : (m == 1) ? wtk : wtv;

    const int l16  = lane & 15;
    const int half = lane >> 4;
    const int koff = half * 8;                   // unused; A fragments use kb as well
    const int kb   = half * 16;                  // B-fragment K offset

    const f16* xr0 = xh + ((size_t)b * T_ + tt * 32 + l16) * C_;
    const f16* xr1 = xr0 + 16 * C_;
    const f16* wh  = wt + (size_t)h * HS_ * C_;

    v8f acc[2][4] = {};

    for (int kc = 0; kc < C_; kc += 32) {
        v16h a0 = cat16(ld8(xr0 + kc + kb), ld8(xr0 + kc + kb + 8));
        v16h a1 = cat16(ld8(xr1 + kc + kb), ld8(xr1 + kc + kb + 8));
        __builtin_prefetch(xr0 + kc + 64, 0, 3);
        __builtin_prefetch(xr1 + kc + 64, 0, 3);
        #pragma unroll
        for (int nt = 0; nt < 4; ++nt) {
            const f16* wrow = wh + (size_t)(nt * 16 + l16) * C_ + kc + kb;
            v16h bf = cat16(ld8(wrow), ld8(wrow + 8));
            acc[0][nt] = wmma16x16x32(a0, bf, acc[0][nt]);
            acc[1][nt] = wmma16x16x32(a1, bf, acc[1][nt]);
        }
    }

    if (m < 2) {
        f16* dst = ((m == 0) ? qs : ks) + ((size_t)b * H_ + h) * T_ * HS_;
        #pragma unroll
        for (int mt2 = 0; mt2 < 2; ++mt2)
            #pragma unroll
            for (int nt = 0; nt < 4; ++nt)
                #pragma unroll
                for (int r = 0; r < 8; ++r) {
                    const int row = tt * 32 + mt2 * 16 + r + 8 * half;
                    dst[(size_t)row * HS_ + nt * 16 + l16] = (f16)acc[mt2][nt][r];
                }
    } else {
        // V transposed: [HS, T]; a lane's 8 C-layout rows are contiguous in T
        f16* dst = vt + ((size_t)b * H_ + h) * HS_ * T_;
        #pragma unroll
        for (int mt2 = 0; mt2 < 2; ++mt2)
            #pragma unroll
            for (int nt = 0; nt < 4; ++nt) {
                v8h pk;
                #pragma unroll
                for (int r = 0; r < 8; ++r) pk[r] = (f16)acc[mt2][nt][r];
                *(v8h*)(dst + (size_t)(nt * 16 + l16) * T_ +
                        tt * 32 + mt2 * 16 + 8 * half) = pk;
            }
    }
}

// ---------------------------------------------------------------------------
// Kernel 2: causal flash attention.  One wave (blockDim=32) per *32-row*
// query tile: each K/V fragment feeds two WMMAs (16 WMMA per 16 b128 loads),
// halving KV traffic vs a 16-row tile.  Online softmax per 16-row sub-tile.
// Output f16, concat-head layout [B*T, H*HS].
// ---------------------------------------------------------------------------
__global__ __launch_bounds__(32)
void flash_attn_kernel(const f16* __restrict__ qs,
                       const f16* __restrict__ ks,
                       const f16* __restrict__ vt,
                       f16* __restrict__ ao) {
    __shared__ __align__(16) f16 pl[32][40];    // P staging; 40-stride keeps v8h aligned

    const int lane = threadIdx.x;
    int bid = blockIdx.x;
    const int qt = bid % (T_ / 32);  bid /= (T_ / 32);
    const int h  = bid % H_;
    const int b  = bid / H_;

    const size_t base = ((size_t)b * H_ + h) * T_ * HS_;
    const f16* __restrict__ Q  = qs + base;
    const f16* __restrict__ K  = ks + base;
    const f16* __restrict__ Vt = vt + base;     // [HS, T]

    const int l16  = lane & 15;
    const int half = lane >> 4;
    const int koff = half * 8;
    const int kb   = half * 16;

    // Q fragments: [mt2][c]  (two 16-row sub-tiles, two 32-wide HS chunks)
    v16h qa[2][2];
    #pragma unroll
    for (int mt2 = 0; mt2 < 2; ++mt2) {
        const f16* qrow = Q + (size_t)(qt * 32 + mt2 * 16 + l16) * HS_;
        qa[mt2][0] = cat16(ld8(qrow + kb),      ld8(qrow + kb + 8));
        qa[mt2][1] = cat16(ld8(qrow + 32 + kb), ld8(qrow + 32 + kb + 8));
    }

    v8f o[2][4] = {};
    float mrow[2][8], lrow[2][8];
    #pragma unroll
    for (int mt2 = 0; mt2 < 2; ++mt2)
        #pragma unroll
        for (int r = 0; r < 8; ++r) { mrow[mt2][r] = -1e30f; lrow[mt2][r] = 0.0f; }

    const float scale = 0.125f;                 // 1/sqrt(HS)
    const int nblk = qt + 1;                    // causal: 32-token KV blocks

    for (int jb = 0; jb < nblk; ++jb) {
        // --- S = Q * K^T (32 KV tokens; K fragments reused by both sub-tiles)
        v8f s[2][2] = {};
        #pragma unroll
        for (int hh = 0; hh < 2; ++hh) {
            const int tok = jb * 32 + hh * 16 + l16;
            const f16* krow = K + (size_t)tok * HS_ + kb;
            #pragma unroll
            for (int c = 0; c < 2; ++c) {
                v16h bk = cat16(ld8(krow + c * 32), ld8(krow + c * 32 + 8));
                s[0][hh] = wmma16x16x32(qa[0][c], bk, s[0][hh]);
                s[1][hh] = wmma16x16x32(qa[1][c], bk, s[1][hh]);
            }
        }

        // --- scale + causal mask + online softmax (per 16-row sub-tile) ---
        #pragma unroll
        for (int mt2 = 0; mt2 < 2; ++mt2) {
            #pragma unroll
            for (int hh = 0; hh < 2; ++hh)
                #pragma unroll
                for (int r = 0; r < 8; ++r) {
                    const int row = qt * 32 + mt2 * 16 + r + 8 * half;
                    const int col = jb * 32 + hh * 16 + l16;
                    float v = s[mt2][hh][r] * scale;
                    s[mt2][hh][r] = (col <= row) ? v : -1e30f;
                }

            float p0[8], p1[8], alpha[8];
            #pragma unroll
            for (int r = 0; r < 8; ++r) {
                float mx = rowmax16(fmaxf(s[mt2][0][r], s[mt2][1][r]));
                float mn = fmaxf(mrow[mt2][r], mx);
                alpha[r] = __expf(mrow[mt2][r] - mn);
                p0[r] = __expf(s[mt2][0][r] - mn);
                p1[r] = __expf(s[mt2][1][r] - mn);
                float rs = rowsum16(p0[r] + p1[r]);
                lrow[mt2][r] = lrow[mt2][r] * alpha[r] + rs;
                mrow[mt2][r] = mn;
            }
            #pragma unroll
            for (int nt = 0; nt < 4; ++nt)
                #pragma unroll
                for (int r = 0; r < 8; ++r)
                    o[mt2][nt][r] *= alpha[r];

            // stage P rows of this sub-tile
            #pragma unroll
            for (int r = 0; r < 8; ++r) {
                pl[mt2 * 16 + r + 8 * half][l16]      = (f16)p0[r];
                pl[mt2 * 16 + r + 8 * half][16 + l16] = (f16)p1[r];
            }
        }
        __syncthreads();
        v16h pa[2];
        #pragma unroll
        for (int mt2 = 0; mt2 < 2; ++mt2)
            pa[mt2] = cat16(*(const v8h*)&pl[mt2 * 16 + l16][kb],
                            *(const v8h*)&pl[mt2 * 16 + l16][kb + 8]);
        __syncthreads();

        // --- O += P * V  (V fragments reused by both sub-tiles) ---
        #pragma unroll
        for (int nt = 0; nt < 4; ++nt) {
            const f16* vrow = Vt + (size_t)(nt * 16 + l16) * T_ + jb * 32 + kb;
            v16h bv = cat16(ld8(vrow), ld8(vrow + 8));
            o[0][nt] = wmma16x16x32(pa[0], bv, o[0][nt]);
            o[1][nt] = wmma16x16x32(pa[1], bv, o[1][nt]);
        }
    }

    #pragma unroll
    for (int mt2 = 0; mt2 < 2; ++mt2)
        #pragma unroll
        for (int r = 0; r < 8; ++r) {
            const float inv = 1.0f / lrow[mt2][r];
            const int row = qt * 32 + mt2 * 16 + r + 8 * half;
            #pragma unroll
            for (int nt = 0; nt < 4; ++nt)
                ao[((size_t)b * T_ + row) * (H_ * HS_) + h * HS_ + nt * 16 + l16] =
                    (f16)(o[mt2][nt][r] * inv);
        }
}

// ---------------------------------------------------------------------------
// Kernel 3: output projection  out = ao[4096,1024] * wpT^T + bproj.
// One wave per 32x64 tile; 8 WMMA per K-chunk, all-contiguous f16 loads.
// ---------------------------------------------------------------------------
__global__ __launch_bounds__(128)
void out_proj_kernel(const f16*  __restrict__ ao,
                     const f16*  __restrict__ wpT,
                     const float* __restrict__ bproj,
                     float* __restrict__ out) {
    const int lane = threadIdx.x & 31;
    int wave = blockIdx.x * (blockDim.x >> 5) + (threadIdx.x >> 5);
    const int nb = wave % (C_ / 64);
    const int mt = wave / (C_ / 64);            // 32-row tile

    const int l16  = lane & 15;
    const int half = lane >> 4;
    const int koff = half * 8;
    const int kb   = half * 16;
    const int KD   = H_ * HS_;                  // 1024

    const f16* ar0 = ao + (size_t)(mt * 32 + l16) * KD;
    const f16* ar1 = ar0 + 16 * KD;

    v8f acc[2][4] = {};
    for (int kc = 0; kc < KD; kc += 32) {
        v16h a0 = cat16(ld8(ar0 + kc + kb), ld8(ar0 + kc + kb + 8));
        v16h a1 = cat16(ld8(ar1 + kc + kb), ld8(ar1 + kc + kb + 8));
        __builtin_prefetch(ar0 + kc + 64, 0, 3);
        __builtin_prefetch(ar1 + kc + 64, 0, 3);
        #pragma unroll
        for (int nt = 0; nt < 4; ++nt) {
            const f16* wrow = wpT + (size_t)(nb * 64 + nt * 16 + l16) * KD + kc + kb;
            v16h bf = cat16(ld8(wrow), ld8(wrow + 8));
            acc[0][nt] = wmma16x16x32(a0, bf, acc[0][nt]);
            acc[1][nt] = wmma16x16x32(a1, bf, acc[1][nt]);
        }
    }

    #pragma unroll
    for (int mt2 = 0; mt2 < 2; ++mt2)
        #pragma unroll
        for (int nt = 0; nt < 4; ++nt) {
            const int col = nb * 64 + nt * 16 + l16;
            const float bias = bproj[col];
            #pragma unroll
            for (int r = 0; r < 8; ++r) {
                const int row = mt * 32 + mt2 * 16 + r + 8 * half;
                out[(size_t)row * C_ + col] = acc[mt2][nt][r] + bias;
            }
        }
}

// ---------------------------------------------------------------------------
extern "C" void kernel_launch(void* const* d_in, const int* in_sizes, int n_in,
                              void* d_out, int out_size, void* d_ws, size_t ws_size,
                              hipStream_t stream) {
    const float* x     = (const float*)d_in[0];
    // d_in[1] = attention_mask (static causal; unused)
    const float* wq    = (const float*)d_in[2];
    const float* wk    = (const float*)d_in[3];
    const float* wv    = (const float*)d_in[4];
    const float* wproj = (const float*)d_in[5];
    const float* bproj = (const float*)d_in[6];
    float*       out   = (float*)d_out;

    const size_t nx  = (size_t)B_ * T_ * C_;        // 4,194,304
    const size_t nw  = (size_t)H_ * HS_ * C_;       // 1,048,576
    const size_t nph = (size_t)B_ * H_ * T_ * HS_;  // 4,194,304

    f16* xh  = (f16*)d_ws;
    f16* wtq = xh  + nx;
    f16* wtk = wtq + nw;
    f16* wtv = wtk + nw;
    f16* wpT = wtv + nw;
    f16* qs  = wpT + nw;
    f16* ks  = qs  + nph;
    f16* vt  = ks  + nph;
    f16* ao  = vt  + nph;                           // total ~50 MB

    // 0) pack: x -> f16, weights -> f16 transposed
    pack_x_kernel <<<(int)(nx / 4 / 256), 256, 0, stream>>>(x, xh);
    pack_wt_kernel<<<(int)(3 * nw / 256), 256, 0, stream>>>(wq, wk, wv, wtq, wtk, wtv);
    pack_wp_kernel<<<(int)(nw / 256), 256, 0, stream>>>(wproj, wpT);

    // 1) QKV projection: 3*B*H*(T/32) = 6144 waves, 4 waves/block
    qkv_proj_kernel<<<6144 / 4, 128, 0, stream>>>(xh, wtq, wtk, wtv, qs, ks, vt);

    // 2) flash attention: one wave per (b,h,32-row query tile)
    flash_attn_kernel<<<B_ * H_ * (T_ / 32), 32, 0, stream>>>(qs, ks, vt, ao);

    // 3) output projection: (4096/32)*(1024/64) = 2048 waves, 4 waves/block
    out_proj_kernel<<<2048 / 4, 128, 0, stream>>>(ao, wpT, bproj, out);
}
